// GenericGAT_10694468567474
// MI455X (gfx1250) — matrix operand order repaired and checked by reading, not compiled, not run on hardware
//
#include <hip/hip_runtime.h>
#include <math.h>

#define N_NODES 50000
#define N_EDGES 800000
#define IN_FEAT 128
#define HID_F   128
#define CLS_F   32
#define NHEAD   4

typedef __attribute__((ext_vector_type(16))) __bf16 v16bf;
typedef __attribute__((ext_vector_type(8)))  __bf16 v8bf;
typedef __attribute__((ext_vector_type(8)))  float  v8f;

// ---------- helpers ----------

static __device__ __forceinline__ __bf16 f2bf(float f) {
  unsigned u = __float_as_uint(f);
  u += 0x7FFFu + ((u >> 16) & 1u);              // round-to-nearest-even
  unsigned short s = (unsigned short)(u >> 16);
  return __builtin_bit_cast(__bf16, s);
}

// order-preserving float<->uint mapping for atomicMax on floats
static __device__ __forceinline__ unsigned f2ord(float f) {
  unsigned u = __float_as_uint(f);
  return (u & 0x80000000u) ? ~u : (u | 0x80000000u);
}
static __device__ __forceinline__ float ord2f(unsigned u) {
  unsigned v = (u & 0x80000000u) ? (u ^ 0x80000000u) : ~u;
  return __uint_as_float(v);
}
#define ORD_NEG_INF 0x007FFFFFu   // f2ord(-inf)

// ---------- conversion kernels (run once per matrix) ----------

__global__ void cvt_bf16(const float* __restrict__ in, __bf16* __restrict__ outb, int n) {
  int i = blockIdx.x * blockDim.x + threadIdx.x;
  if (i < n) outb[i] = f2bf(in[i]);
}

// Bt[k*A + n] = bf16(W[n*F + k])  -> exact B-fragment memory layout
__global__ void wt_bf16(const float* __restrict__ W, __bf16* __restrict__ Bt,
                        int A, int F) {
  int i = blockIdx.x * blockDim.x + threadIdx.x;
  if (i >= A * F) return;
  int k = i / A, n = i % A;
  Bt[i] = f2bf(W[(size_t)n * F + k]);
}

// ---------- WMMA GEMM: out[M,A] = featb[M,F](bf16) @ Bt[F,A](bf16) + bias ----------
// One wave computes a 16 x A strip: A-fragment loaded once per K-step and
// reused across all A/16 column tiles (8 WMMAs per K-step for A=128).
// A-frag (ISA 7.12.2, 16-bit A 16x32): lane -> row m (mod 16), half selects
// K-groups {koff..koff+7, 16+koff..}. B-frag: lane -> col n (mod 16), same K-groups.
template<int A, int F>
__global__ void __launch_bounds__(256)
gemm_bf16_t(const __bf16* __restrict__ featb, const __bf16* __restrict__ Bt,
            const float* __restrict__ bias, float* __restrict__ out, int M) {
  constexpr int NT = A / 16;
  const int lane = threadIdx.x & 31;
  const int wave = threadIdx.x >> 5;
  const int tm   = blockIdx.x * 8 + wave;
  if (tm >= (M >> 4)) return;                  // wave-uniform: EXEC stays full
  const int m0   = tm << 4;
  const int mrow = m0 + (lane & 15);
  const int koff = (lane >> 4) << 3;           // 0 or 8

  v8f c[NT];
#pragma unroll
  for (int t = 0; t < NT; ++t) c[t] = (v8f){};

#pragma unroll
  for (int kk = 0; kk < F; kk += 32) {
    const __bf16* fr = featb + (size_t)mrow * F + kk + koff;
    v8bf lo = *(const v8bf*)(fr);              // K = kk+koff .. +7
    v8bf hi = *(const v8bf*)(fr + 16);         // K = kk+16+koff .. +7
    v16bf a;
#pragma unroll
    for (int j = 0; j < 8; ++j) { a[j] = lo[j]; a[8 + j] = hi[j]; }
    const __bf16* br = Bt + (size_t)(kk + koff) * A + (lane & 15);
#pragma unroll
    for (int t = 0; t < NT; ++t) {
      v16bf b;
#pragma unroll
      for (int j = 0; j < 8; ++j) { b[j] = br[(size_t)j * A + t * 16]; b[8 + j] = br[(size_t)(16 + j) * A + t * 16]; }
      c[t] = __builtin_amdgcn_wmma_f32_16x16x32_bf16(false, a, false, b,
                                                     (short)0, c[t], false, false);
    }
  }

  const int nl = lane & 15;
  const int mb = m0 + ((lane >> 4) << 3);
#pragma unroll
  for (int t = 0; t < NT; ++t) {
    const int n  = t * 16 + nl;
    const float bn = bias[n];
#pragma unroll
    for (int r = 0; r < 8; ++r)
      out[(size_t)(mb + r) * A + n] = c[t][r] + bn;
  }
}

// ---------- fills ----------
__global__ void fill_f32(float* __restrict__ p, float v, int n) {
  int i = blockIdx.x * blockDim.x + threadIdx.x;
  if (i < n) p[i] = v;
}
__global__ void fill_u32(unsigned* __restrict__ p, unsigned v, int n) {
  int i = blockIdx.x * blockDim.x + threadIdx.x;
  if (i < n) p[i] = v;
}

// ---------- edge pass 1: scores + fused segment-max (keyed on src) ----------
__global__ void edge_scores(const float* __restrict__ q, const float* __restrict__ k,
                            const long long* __restrict__ edge,
                            float* __restrict__ sc, unsigned* __restrict__ mbuf,
                            int A, int dk, float scale) {
  int idx = blockIdx.x * blockDim.x + threadIdx.x;
  if (idx >= N_EDGES * NHEAD) return;
  int e = idx >> 2, h = idx & 3;                // H == 4
  int src = (int)edge[e];
  int dst = (int)edge[N_EDGES + e];
  const float4* qr = (const float4*)(q + (size_t)src * A + h * dk);
  const float4* kr = (const float4*)(k + (size_t)dst * A + h * dk);
  float acc = 0.f;
  for (int j = 0; j < (dk >> 2); ++j) {
    float4 a = qr[j], b = kr[j];
    acc += a.x * b.x + a.y * b.y + a.z * b.z + a.w * b.w;
  }
  float s = acc * scale;
  sc[idx] = s;
  atomicMax(&mbuf[src * NHEAD + h], f2ord(s));
}

// ---------- edge pass 2: exp(s - m) + segment-sum ----------
__global__ void edge_expsum(const long long* __restrict__ edge,
                            float* __restrict__ sc,
                            const unsigned* __restrict__ mbuf,
                            float* __restrict__ denom) {
  int idx = blockIdx.x * blockDim.x + threadIdx.x;
  if (idx >= N_EDGES * NHEAD) return;
  int e = idx >> 2, h = idx & 3;
  int src = (int)edge[e];
  float m = ord2f(mbuf[src * NHEAD + h]);
  float v = __expf(sc[idx] - m);
  sc[idx] = v;
  atomicAdd(&denom[src * NHEAD + h], v);
}

// ---------- edge pass 3: out[src] += beta * attn * v[dst] ----------
__global__ void edge_scatter(const float* __restrict__ vmat,
                             const long long* __restrict__ edge,
                             const float* __restrict__ sc,
                             const float* __restrict__ denom,
                             float* __restrict__ out,
                             int A, int dkshift, float beta) {
  long long idx = (long long)blockIdx.x * blockDim.x + threadIdx.x;
  long long tot = (long long)N_EDGES * A;
  if (idx >= tot) return;
  int e = (int)(idx / A);
  int c = (int)(idx % A);
  int h = c >> dkshift;
  int src = (int)edge[e];
  int dst = (int)edge[N_EDGES + e];
  float attn = sc[e * NHEAD + h] / (denom[src * NHEAD + h] + 1e-16f);
  atomicAdd(&out[(size_t)src * A + c],
            beta * attn * vmat[(size_t)dst * A + c]);
}

// ---------- pointwise ----------
__global__ void elu_inplace(float* __restrict__ p, int n) {
  int i = blockIdx.x * blockDim.x + threadIdx.x;
  if (i < n) { float x = p[i]; p[i] = x > 0.f ? x : expm1f(x); }
}

// wave32: one wave per node, 32 lanes == 32 classes
__global__ void __launch_bounds__(256)
log_softmax32(const float* __restrict__ logits, float* __restrict__ out) {
  int lane = threadIdx.x & 31;
  int row  = (blockIdx.x * blockDim.x + threadIdx.x) >> 5;
  if (row >= N_NODES) return;
  float x  = logits[(size_t)row * CLS_F + lane];
  float mx = x;
  for (int o = 16; o; o >>= 1) mx = fmaxf(mx, __shfl_xor(mx, o, 32));
  float s = __expf(x - mx);
  for (int o = 16; o; o >>= 1) s += __shfl_xor(s, o, 32);
  out[(size_t)row * CLS_F + lane] = (x - mx) - logf(s);
}

// ---------- driver ----------
extern "C" void kernel_launch(void* const* d_in, const int* in_sizes, int n_in,
                              void* d_out, int out_size, void* d_ws, size_t ws_size,
                              hipStream_t stream) {
  const float* x   = (const float*)d_in[0];
  const float* Wq0 = (const float*)d_in[1];
  const float* bq0 = (const float*)d_in[2];
  const float* Wk0 = (const float*)d_in[3];
  const float* bk0 = (const float*)d_in[4];
  const float* Wv0 = (const float*)d_in[5];
  const float* bv0 = (const float*)d_in[6];
  const float* Wq1 = (const float*)d_in[7];
  const float* bq1 = (const float*)d_in[8];
  const float* Wk1 = (const float*)d_in[9];
  const float* bk1 = (const float*)d_in[10];
  const float* Wv1 = (const float*)d_in[11];
  const float* bv1 = (const float*)d_in[12];
  const long long* eL0[2] = { (const long long*)d_in[13], (const long long*)d_in[14] };
  const long long* eL1[2] = { (const long long*)d_in[15], (const long long*)d_in[16] };

  // workspace layout; hot set stays L2-resident (~136 MB < 192 MB L2)
  float*    q      = (float*)d_ws;                          // N*128 f32
  float*    k      = q + (size_t)N_NODES * HID_F;
  float*    v      = k + (size_t)N_NODES * HID_F;
  float*    h      = v + (size_t)N_NODES * HID_F;
  float*    sc     = h + (size_t)N_NODES * HID_F;           // E*4 f32
  unsigned* mbuf   = (unsigned*)(sc + (size_t)N_EDGES * NHEAD);
  float*    denom  = (float*)(mbuf + (size_t)N_NODES * NHEAD);
  float*    logits = denom + (size_t)N_NODES * NHEAD;       // N*32 f32
  __bf16*   featb  = (__bf16*)(logits + (size_t)N_NODES * CLS_F); // N*128 bf16
  __bf16*   Btq    = featb + (size_t)N_NODES * HID_F;       // 128*128 bf16 each
  __bf16*   Btk    = Btq + (size_t)HID_F * HID_F;
  __bf16*   Btv    = Btk + (size_t)HID_F * HID_F;

  const int nh  = N_NODES * NHEAD;
  const int nEH = N_EDGES * NHEAD;
  const int bEH = (nEH + 255) / 256;
  const int bNH = (nh + 255) / 256;
  const int gM  = ((N_NODES / 16) + 7) / 8;   // row-strip blocks, 8 waves/block

  auto run_hop = [&](int A, int dk, int dkshift,
                     const float* Wq, const float* bq,
                     const float* Wk, const float* bk,
                     const float* Wv, const float* bv,
                     const long long* edge, float* outacc, float beta) {
    const int wElems = A * HID_F;
    const int bW = (wElems + 255) / 256;
    wt_bf16<<<bW, 256, 0, stream>>>(Wq, Btq, A, HID_F);
    wt_bf16<<<bW, 256, 0, stream>>>(Wk, Btk, A, HID_F);
    wt_bf16<<<bW, 256, 0, stream>>>(Wv, Btv, A, HID_F);
    if (A == 128) {
      gemm_bf16_t<128, 128><<<gM, 256, 0, stream>>>(featb, Btq, bq, q, N_NODES);
      gemm_bf16_t<128, 128><<<gM, 256, 0, stream>>>(featb, Btk, bk, k, N_NODES);
      gemm_bf16_t<128, 128><<<gM, 256, 0, stream>>>(featb, Btv, bv, v, N_NODES);
    } else {
      gemm_bf16_t<32, 128><<<gM, 256, 0, stream>>>(featb, Btq, bq, q, N_NODES);
      gemm_bf16_t<32, 128><<<gM, 256, 0, stream>>>(featb, Btk, bk, k, N_NODES);
      gemm_bf16_t<32, 128><<<gM, 256, 0, stream>>>(featb, Btv, bv, v, N_NODES);
    }
    fill_u32<<<bNH, 256, 0, stream>>>(mbuf, ORD_NEG_INF, nh);
    fill_f32<<<bNH, 256, 0, stream>>>(denom, 0.f, nh);
    const float scale = 1.f / sqrtf((float)dk);
    edge_scores<<<bEH, 256, 0, stream>>>(q, k, edge, sc, mbuf, A, dk, scale);
    edge_expsum<<<bEH, 256, 0, stream>>>(edge, sc, mbuf, denom);
    const long long tot = (long long)N_EDGES * A;
    const int bSc = (int)((tot + 255) / 256);
    edge_scatter<<<bSc, 256, 0, stream>>>(v, edge, sc, denom, outacc, A, dkshift, beta);
  };

  const int nHF = N_NODES * HID_F;
  const int bHF = (nHF + 255) / 256;

  // ----- layer 0: feat = x, A=128, dk=32 -----
  {
    cvt_bf16<<<bHF, 256, 0, stream>>>(x, featb, nHF);   // IN_FEAT == HID_F == 128
    fill_f32<<<bHF, 256, 0, stream>>>(h, 0.f, nHF);
    const size_t wOff = (size_t)HID_F * IN_FEAT, bOff = HID_F;
    run_hop(HID_F, 32, 5, Wq0, bq0, Wk0, bk0, Wv0, bv0, eL0[0], h, 1.0f);
    run_hop(HID_F, 32, 5, Wq0 + wOff, bq0 + bOff, Wk0 + wOff, bk0 + bOff,
            Wv0 + wOff, bv0 + bOff, eL0[1], h, 0.5f);
    elu_inplace<<<bHF, 256, 0, stream>>>(h, nHF);
  }

  // ----- layer 1: feat = h, A=32, dk=8 -----
  {
    const int nCF = N_NODES * CLS_F;
    cvt_bf16<<<bHF, 256, 0, stream>>>(h, featb, nHF);
    fill_f32<<<(nCF + 255) / 256, 256, 0, stream>>>(logits, 0.f, nCF);
    const size_t wOff = (size_t)CLS_F * HID_F, bOff = CLS_F;
    run_hop(CLS_F, 8, 3, Wq1, bq1, Wk1, bk1, Wv1, bv1, eL1[0], logits, 1.0f);
    run_hop(CLS_F, 8, 3, Wq1 + wOff, bq1 + bOff, Wk1 + wOff, bk1 + bOff,
            Wv1 + wOff, bv1 + bOff, eL1[1], logits, 0.5f);
  }

  // ----- log_softmax -----
  log_softmax32<<<(N_NODES + 7) / 8, 256, 0, stream>>>(logits, (float*)d_out);
}
